// MultHeadAttention_32624571580737
// MI455X (gfx1250) — compile-verified, matrix operand order checked by reading, not run on hardware
//
#include <hip/hip_runtime.h>
#include <hip/hip_bf16.h>

// ---------------------------------------------------------------------------
// MHA forward for B=4, T=2048, D=1024, H=16, HD=64 on gfx1250 (MI455X).
// bf16 WMMA (v_wmma_f32_16x16x32_bf16) everywhere, fp32 accumulate.
// ---------------------------------------------------------------------------

typedef __attribute__((ext_vector_type(16))) __bf16 v16bf;
typedef __attribute__((ext_vector_type(8)))  float  v8f;
typedef __attribute__((ext_vector_type(4)))  __bf16 v4bf;

union Frag16 { v16bf v; uint4 q[2]; };

__device__ __forceinline__ v8f wmma_bf16(v16bf a, v16bf b, v8f c) {
  // 8 args: (neg_a, A, neg_b, B, c_mod, C, reuse_a, reuse_b)
  return __builtin_amdgcn_wmma_f32_16x16x32_bf16(false, a, false, b,
                                                 (short)0, c, false, false);
}

// row-wise reductions across a 16-lane half of the wave (wave32; xor masks
// 1,2,4,8 stay inside each half)
__device__ __forceinline__ float red16_max(float v) {
  v = fmaxf(v, __shfl_xor(v, 1));
  v = fmaxf(v, __shfl_xor(v, 2));
  v = fmaxf(v, __shfl_xor(v, 4));
  v = fmaxf(v, __shfl_xor(v, 8));
  return v;
}
__device__ __forceinline__ float red16_sum(float v) {
  v += __shfl_xor(v, 1);
  v += __shfl_xor(v, 2);
  v += __shfl_xor(v, 4);
  v += __shfl_xor(v, 8);
  return v;
}

// ---------------------------------------------------------------------------
// fp32 -> bf16 conversion (vectorized, n % 4 == 0)
// ---------------------------------------------------------------------------
__global__ __launch_bounds__(256) void cvt_f32_bf16(const float* __restrict__ x,
                                                    __bf16* __restrict__ y, int n) {
  int i = (blockIdx.x * blockDim.x + threadIdx.x) * 4;
  if (i >= n) return;
  const float4 f = *(const float4*)(x + i);
  v4bf o;
  o[0] = (__bf16)f.x; o[1] = (__bf16)f.y; o[2] = (__bf16)f.z; o[3] = (__bf16)f.w;
  *(v4bf*)(y + i) = o;
}

// ---------------------------------------------------------------------------
// GEMM: Y = X @ W^T + bias.  X:[8192,1024] bf16 row-major, W:[1024,1024] bf16
// (torch Linear weight, [out,in]).  Block tile 128x128, 8 waves, wave = 32x64.
// MODE 0: store bf16 Q-heads [bh][t][64], pre-scaled by 1/sqrt(64)
// MODE 1: store bf16 K-heads [bh][t][64]
// MODE 2: store bf16 V-heads transposed [bh][64][t]
// MODE 3: store fp32 plain [m][n] (output projection)
// ---------------------------------------------------------------------------
template <int MODE>
__global__ __launch_bounds__(256) void gemm_qkv(const __bf16* __restrict__ X,
                                                const __bf16* __restrict__ W,
                                                const float* __restrict__ bias,
                                                void* __restrict__ out) {
  constexpr int K = 1024;
  const int lane = threadIdx.x & 31;
  const int wave = threadIdx.x >> 5;
  const int kh   = lane >> 4;   // which K-half of the fragment this lane holds
  const int ln   = lane & 15;
  const int m0 = blockIdx.x * 128 + (wave & 3) * 32;
  const int n0 = blockIdx.y * 128 + (wave >> 2) * 64;

  const __bf16* xr0 = X + (size_t)(m0 + ln) * K;
  const __bf16* xr1 = X + (size_t)(m0 + 16 + ln) * K;
  const __bf16* wr0 = W + (size_t)(n0 + 0  + ln) * K;
  const __bf16* wr1 = W + (size_t)(n0 + 16 + ln) * K;
  const __bf16* wr2 = W + (size_t)(n0 + 32 + ln) * K;
  const __bf16* wr3 = W + (size_t)(n0 + 48 + ln) * K;
  const __bf16* wr[4] = {wr0, wr1, wr2, wr3};

  v8f acc[2][4];
#pragma unroll
  for (int i = 0; i < 2; ++i)
#pragma unroll
    for (int j = 0; j < 4; ++j)
#pragma unroll
      for (int r = 0; r < 8; ++r) acc[i][j][r] = 0.f;

  for (int k0 = 0; k0 < K; k0 += 32) {
    const int ka = k0 + 8 * kh;  // 16-bit A-layout: lanes 0-15 K=0..7/16..23
    Frag16 a0, a1;
    a0.q[0] = *(const uint4*)(xr0 + ka);
    a0.q[1] = *(const uint4*)(xr0 + ka + 16);
    a1.q[0] = *(const uint4*)(xr1 + ka);
    a1.q[1] = *(const uint4*)(xr1 + ka + 16);
    __builtin_prefetch(xr0 + ka + 32, 0, 0);   // global_prefetch_b8
    __builtin_prefetch(xr1 + ka + 32, 0, 0);
#pragma unroll
    for (int ni = 0; ni < 4; ++ni) {
      Frag16 bf;
      bf.q[0] = *(const uint4*)(wr[ni] + ka);
      bf.q[1] = *(const uint4*)(wr[ni] + ka + 16);
      acc[0][ni] = wmma_bf16(a0.v, bf.v, acc[0][ni]);
      acc[1][ni] = wmma_bf16(a1.v, bf.v, acc[1][ni]);
    }
  }

  float bn[4];
#pragma unroll
  for (int ni = 0; ni < 4; ++ni) bn[ni] = bias[n0 + ni * 16 + ln];

#pragma unroll
  for (int mi = 0; mi < 2; ++mi)
#pragma unroll
    for (int ni = 0; ni < 4; ++ni)
#pragma unroll
      for (int r = 0; r < 8; ++r) {
        const int m = m0 + mi * 16 + r + 8 * kh;  // C layout: vgpr r = row r(+8)
        const int n = n0 + ni * 16 + ln;
        float val = acc[mi][ni][r] + bn[ni];
        if constexpr (MODE == 3) {
          ((float*)out)[(size_t)m * 1024 + n] = val;
        } else {
          if constexpr (MODE == 0) val *= 0.125f;  // 1/sqrt(HD)
          const int b = m >> 11, t = m & 2047;     // T = 2048
          const int h = n >> 6,  d = n & 63;       // HD = 64
          size_t idx;
          if constexpr (MODE == 2)
            idx = (size_t)(b * 16 + h) * (64 * 2048) + (size_t)d * 2048 + t;
          else
            idx = (size_t)(b * 16 + h) * (2048 * 64) + (size_t)t * 64 + d;
          ((__bf16*)out)[idx] = (__bf16)val;
        }
      }
}

// ---------------------------------------------------------------------------
// One 64-key streaming block of causal flash attention for one wave
// (16 query rows).  MASK=false for interior (fully-unmasked) blocks; the
// diagonal-straddling block is always the last one, so only it pays for
// the per-element compare/select.
// ---------------------------------------------------------------------------
template <bool MASK>
__device__ __forceinline__ void attn_block(
    int j0, int i0, int kh, int ln,
    const __bf16* __restrict__ Kp, const __bf16* __restrict__ Vp,
    const Frag16 (&qa)[2], __bf16* pb /* this wave's 16x64 LDS tile */,
    v8f (&o)[4], float (&mrow)[8], float (&lrow)[8]) {
  constexpr int T = 2048, HD = 64;

  // ---- S = Q · K^T for 64 keys (four 16-col fragments, 8 WMMAs) ----
  v8f s[4];
#pragma unroll
  for (int cb = 0; cb < 4; ++cb) {
#pragma unroll
    for (int r = 0; r < 8; ++r) s[cb][r] = 0.f;
    const __bf16* krow = Kp + (size_t)(j0 + cb * 16 + ln) * HD;
#pragma unroll
    for (int dt = 0; dt < 2; ++dt) {
      Frag16 kb;
      const int base = dt * 32 + 8 * kh;
      kb.q[0] = *(const uint4*)(krow + base);
      kb.q[1] = *(const uint4*)(krow + base + 16);
      s[cb] = wmma_bf16(qa[dt].v, kb.v, s[cb]);
    }
  }

  // ---- online softmax update (one reduction chain per row per 64 keys) ----
#pragma unroll
  for (int r = 0; r < 8; ++r) {
    float v[4];
#pragma unroll
    for (int cb = 0; cb < 4; ++cb) {
      v[cb] = s[cb][r];
      if (MASK) {
        const int i  = i0 + r + 8 * kh;
        const int jc = j0 + cb * 16 + ln;
        v[cb] = (jc <= i) ? v[cb] : -10000.0f;
      }
    }
    float mx = fmaxf(fmaxf(v[0], v[1]), fmaxf(v[2], v[3]));
    mx = red16_max(mx);
    const float mnew = fmaxf(mrow[r], mx);
    const float corr = __expf(mrow[r] - mnew);
    mrow[r] = mnew;
    float p[4], ps = 0.f;
#pragma unroll
    for (int cb = 0; cb < 4; ++cb) { p[cb] = __expf(v[cb] - mnew); ps += p[cb]; }
    lrow[r] = lrow[r] * corr + red16_sum(ps);
#pragma unroll
    for (int dt = 0; dt < 4; ++dt) o[dt][r] *= corr;
    const int row = r + 8 * kh;   // C-layout -> LDS (16x64 bf16 row-major)
#pragma unroll
    for (int cb = 0; cb < 4; ++cb) pb[row * 64 + cb * 16 + ln] = (__bf16)p[cb];
  }

  // ---- re-read P as two A-fragments (same wave; LDS ops are in-order) ----
  Frag16 pa[2];
#pragma unroll
  for (int js = 0; js < 2; ++js) {
    const int base = ln * 64 + js * 32 + 8 * kh;
    pa[js].q[0] = *(const uint4*)(pb + base);
    pa[js].q[1] = *(const uint4*)(pb + base + 16);
  }

  // ---- O += P · V (64 keys, 4 output d-tiles, 8 WMMAs) ----
#pragma unroll
  for (int dt = 0; dt < 4; ++dt) {
    const __bf16* vrow = Vp + (size_t)(dt * 16 + ln) * T + j0;
#pragma unroll
    for (int js = 0; js < 2; ++js) {
      Frag16 vb;
      vb.q[0] = *(const uint4*)(vrow + js * 32 + 8 * kh);
      vb.q[1] = *(const uint4*)(vrow + js * 32 + 16 + 8 * kh);
      o[dt] = wmma_bf16(pa[js].v, vb.v, o[dt]);
    }
  }
}

// ---------------------------------------------------------------------------
// Causal flash attention.  grid = (T/64, B*H), block = 128 (4 waves).
// Each wave owns 16 query rows; streams 64-key blocks with online softmax.
// Qh pre-scaled; Vt is [bh][64][T] so the PV B-operand reads contiguously.
// Output Ao is bf16 [B,T,D] (heads re-interleaved) for the final projection.
// ---------------------------------------------------------------------------
__global__ __launch_bounds__(128) void attn_fwd(const __bf16* __restrict__ Qh,
                                                const __bf16* __restrict__ Kh,
                                                const __bf16* __restrict__ Vt,
                                                __bf16* __restrict__ Ao) {
  constexpr int T = 2048, HD = 64, Dm = 1024;
  __shared__ __align__(16) __bf16 pbuf[4][16 * 64];  // 2 KB per wave

  const int lane = threadIdx.x & 31;
  const int wave = threadIdx.x >> 5;
  const int kh   = lane >> 4;
  const int ln   = lane & 15;
  const int bh   = blockIdx.y;                 // b*16 + h
  const int i0   = blockIdx.x * 64 + wave * 16;

  const __bf16* Qp = Qh + (size_t)bh * T * HD;
  const __bf16* Kp = Kh + (size_t)bh * T * HD;
  const __bf16* Vp = Vt + (size_t)bh * HD * T;
  __bf16* pb = &pbuf[wave][0];

  // Q A-fragments for d = 0..31 and 32..63 (rows i0+ln)
  Frag16 qa[2];
  {
    const __bf16* qrow = Qp + (size_t)(i0 + ln) * HD;
#pragma unroll
    for (int dt = 0; dt < 2; ++dt) {
      const int base = dt * 32 + 8 * kh;
      qa[dt].q[0] = *(const uint4*)(qrow + base);
      qa[dt].q[1] = *(const uint4*)(qrow + base + 16);
    }
  }

  v8f o[4];
  float mrow[8], lrow[8];
#pragma unroll
  for (int dt = 0; dt < 4; ++dt)
#pragma unroll
    for (int r = 0; r < 8; ++r) o[dt][r] = 0.f;
#pragma unroll
  for (int r = 0; r < 8; ++r) { mrow[r] = -1e30f; lrow[r] = 0.f; }

  // Only the final 64-key block can straddle the diagonal for this wave's
  // rows (i0..i0+15): interior blocks need no per-element mask.
  const int jlast = ((i0 + 15) >> 6) << 6;
  for (int j0 = 0; j0 < jlast; j0 += 64)
    attn_block<false>(j0, i0, kh, ln, Kp, Vp, qa, pb, o, mrow, lrow);
  attn_block<true>(jlast, i0, kh, ln, Kp, Vp, qa, pb, o, mrow, lrow);

  // ---- normalize and store to [B,T,D] bf16 (heads interleaved) ----
  const int b = bh >> 4, h = bh & 15;
#pragma unroll
  for (int r = 0; r < 8; ++r) {
    const float inv = 1.0f / lrow[r];
    const int t = i0 + r + 8 * kh;
    const size_t base = ((size_t)(b * T + t)) * Dm + h * 64 + ln;
#pragma unroll
    for (int dt = 0; dt < 4; ++dt)
      Ao[base + dt * 16] = (__bf16)(o[dt][r] * inv);
  }
}

// ---------------------------------------------------------------------------
extern "C" void kernel_launch(void* const* d_in, const int* in_sizes, int n_in,
                              void* d_out, int out_size, void* d_ws, size_t ws_size,
                              hipStream_t stream) {
  (void)in_sizes; (void)n_in; (void)out_size; (void)ws_size;
  const float* q  = (const float*)d_in[0];
  const float* k  = (const float*)d_in[1];
  const float* v  = (const float*)d_in[2];
  // d_in[3] = mask (tril) — causality applied analytically in attn_fwd
  const float* Wq = (const float*)d_in[4];
  const float* bq = (const float*)d_in[5];
  const float* Wk = (const float*)d_in[6];
  const float* bk = (const float*)d_in[7];
  const float* Wv = (const float*)d_in[8];
  const float* bv = (const float*)d_in[9];
  const float* Wo = (const float*)d_in[10];
  const float* bo = (const float*)d_in[11];

  char* ws = (char*)d_ws;
  const size_t MB = 1ull << 20;
  __bf16* qbf = (__bf16*)(ws + 0 * MB);     // 16 MiB each activation
  __bf16* kbf = (__bf16*)(ws + 16 * MB);
  __bf16* vbf = (__bf16*)(ws + 32 * MB);
  __bf16* wqb = (__bf16*)(ws + 48 * MB);    // 2 MiB each weight
  __bf16* wkb = (__bf16*)(ws + 50 * MB);
  __bf16* wvb = (__bf16*)(ws + 52 * MB);
  __bf16* wob = (__bf16*)(ws + 54 * MB);
  __bf16* Qh  = (__bf16*)(ws + 56 * MB);    // [bh][t][64], pre-scaled
  __bf16* Kh  = (__bf16*)(ws + 72 * MB);    // [bh][t][64]
  __bf16* Vt  = (__bf16*)(ws + 88 * MB);    // [bh][64][t]
  __bf16* Ao  = (__bf16*)(ws + 104 * MB);   // [B,T,D] bf16

  const int nAct = 4 * 2048 * 1024;  // 8388608
  const int nW   = 1024 * 1024;
  cvt_f32_bf16<<<nAct / 1024, 256, 0, stream>>>(q, qbf, nAct);
  cvt_f32_bf16<<<nAct / 1024, 256, 0, stream>>>(k, kbf, nAct);
  cvt_f32_bf16<<<nAct / 1024, 256, 0, stream>>>(v, vbf, nAct);
  cvt_f32_bf16<<<nW / 1024, 256, 0, stream>>>(Wq, wqb, nW);
  cvt_f32_bf16<<<nW / 1024, 256, 0, stream>>>(Wk, wkb, nW);
  cvt_f32_bf16<<<nW / 1024, 256, 0, stream>>>(Wv, wvb, nW);
  cvt_f32_bf16<<<nW / 1024, 256, 0, stream>>>(Wo, wob, nW);

  const dim3 gg(64, 8), gb(256);  // M/128 x N/128 tiles
  gemm_qkv<0><<<gg, gb, 0, stream>>>(qbf, wqb, bq, Qh);
  gemm_qkv<1><<<gg, gb, 0, stream>>>(kbf, wkb, bk, Kh);
  gemm_qkv<2><<<gg, gb, 0, stream>>>(vbf, wvb, bv, Vt);

  attn_fwd<<<dim3(32, 64), 128, 0, stream>>>(Qh, Kh, Vt, Ao);

  gemm_qkv<3><<<gg, gb, 0, stream>>>(Ao, wob, bo, d_out);
}
